// RwkvSelfAttention_78426102825333
// MI455X (gfx1250) — matrix operand order repaired and checked by reading, not compiled, not run on hardware
//
#include <hip/hip_runtime.h>
#include <hip/hip_bf16.h>
#include <cstdint>
#include <cstddef>

// ---------------- problem constants (B,T,H from the reference) --------------
constexpr int Bsz = 2;
constexpr int Tsz = 2048;
constexpr int Hsz = 2048;
constexpr int Mrows = Bsz * Tsz;   // 4096 = GEMM M
// All four GEMMs are [M=4096, N=2048, K=2048], row-major, K contiguous on both
// operands (einsum 'bth,ah->bta' and 'bta,ha->bth').

// ---------------- vector types ---------------------------------------------
typedef __attribute__((ext_vector_type(16))) __bf16  bf16x16;
typedef __attribute__((ext_vector_type(4)))  __bf16  bf16x4;
typedef __attribute__((ext_vector_type(8)))  float   f32x8;
typedef __attribute__((ext_vector_type(4)))  unsigned int u32x4;

union FragAB {           // 32 bytes: one bf16 WMMA operand fragment per lane
    u32x4   u[2];
    bf16x16 v;
};

// ---------------- kernel 1: time-shift + lerp, fp32 -> bf16 -----------------
__global__ void prep_mix(const float* __restrict__ hidden,
                         const float* __restrict__ tmk,
                         const float* __restrict__ tmv,
                         const float* __restrict__ tmr,
                         __bf16* __restrict__ kin,
                         __bf16* __restrict__ vin,
                         __bf16* __restrict__ rin)
{
    const size_t tid = (size_t)blockIdx.x * blockDim.x + threadIdx.x;
    const size_t h4  = Hsz / 4;
    const size_t m   = tid / h4;            // row in [0, M)
    const int    h   = (int)(tid % h4) * 4; // 4 contiguous channels
    const int    t   = (int)(m % Tsz);

    const float4 hv = *(const float4*)(hidden + m * Hsz + h);
    float4 sv = make_float4(0.f, 0.f, 0.f, 0.f);
    if (t > 0) sv = *(const float4*)(hidden + (m - 1) * Hsz + h);

    const float4 mk = *(const float4*)(tmk + h);
    const float4 mv = *(const float4*)(tmv + h);
    const float4 mr = *(const float4*)(tmr + h);

    bf16x4 ko, vo, ro;
    ko[0] = (__bf16)(hv.x * mk.x + sv.x * (1.f - mk.x));
    ko[1] = (__bf16)(hv.y * mk.y + sv.y * (1.f - mk.y));
    ko[2] = (__bf16)(hv.z * mk.z + sv.z * (1.f - mk.z));
    ko[3] = (__bf16)(hv.w * mk.w + sv.w * (1.f - mk.w));
    vo[0] = (__bf16)(hv.x * mv.x + sv.x * (1.f - mv.x));
    vo[1] = (__bf16)(hv.y * mv.y + sv.y * (1.f - mv.y));
    vo[2] = (__bf16)(hv.z * mv.z + sv.z * (1.f - mv.z));
    vo[3] = (__bf16)(hv.w * mv.w + sv.w * (1.f - mv.w));
    ro[0] = (__bf16)(hv.x * mr.x + sv.x * (1.f - mr.x));
    ro[1] = (__bf16)(hv.y * mr.y + sv.y * (1.f - mr.y));
    ro[2] = (__bf16)(hv.z * mr.z + sv.z * (1.f - mr.z));
    ro[3] = (__bf16)(hv.w * mr.w + sv.w * (1.f - mr.w));

    *(bf16x4*)(kin + m * Hsz + h) = ko;
    *(bf16x4*)(vin + m * Hsz + h) = vo;
    *(bf16x4*)(rin + m * Hsz + h) = ro;
}

// ---------------- kernel 2: fp32 -> bf16 weight conversion ------------------
__global__ void cvt_f32_bf16(const float* __restrict__ src,
                             __bf16* __restrict__ dst)
{
    const size_t i = (size_t)blockIdx.x * blockDim.x + threadIdx.x;
    const float4 v = ((const float4*)src)[i];
    bf16x4 o;
    o[0] = (__bf16)v.x; o[1] = (__bf16)v.y; o[2] = (__bf16)v.z; o[3] = (__bf16)v.w;
    ((bf16x4*)dst)[i] = o;
}

// ---------------- kernel 3: LDS-staged bf16 WMMA GEMM -----------------------
// C[M,N] = A[M,K] * Bw[N,K]^T. Workgroup tile 128x128, 8 waves in 2x4, each
// wave 64x32 (4x2 WMMA tiles), K staged in 64-wide double-buffered LDS tiles
// filled by GLOBAL_LOAD_ASYNC_TO_LDS_B128 (ASYNCcnt), consumed by ds_load_b128.
// Rows padded to 72 elements (144 B = 36 banks) to spread fragment reads
// across LDS banks.
constexpr int KSTEP = 64;
constexpr int LDAP  = 72;                 // padded LDS row length (elements)
constexpr int TILE_ELEMS = 128 * LDAP;    // one matrix, one buffer

__global__ __launch_bounds__(256)
void gemm_bf16(const __bf16* __restrict__ A,   // [M,K] row-major
               const __bf16* __restrict__ Bw,  // [N,K] row-major (K contig)
               float* __restrict__ C,          // [M,N] row-major
               int act)                        // 0 = none, 1 = sigmoid
{
    constexpr int Kdim = Hsz;   // 2048
    constexpr int Ndim = Hsz;   // 2048
    constexpr int NS   = Kdim / KSTEP;   // 32 stages

    __shared__ __bf16 smem[2 * 2 * TILE_ELEMS];   // [buf][A|B][128][72] = 72 KB

    const int tid  = threadIdx.x;
    const int lane = tid & 31;
    const int wave = tid >> 5;           // 0..7
    const int lr   = lane & 15;          // row (A) / col (B) within 16
    const int lh   = lane >> 4;          // half-wave K sub-chunk select

    const int nBlk = blockIdx.x * 128;
    const int mBlk = blockIdx.y * 128;
    const int mW   = (wave >> 2) * 64;   // wave M base within block tile
    const int nW   = (wave & 3) * 32;    // wave N base within block tile

    // --- async stage: copy 128x64 bf16 of A and of B into LDS buffer `buf`.
    // 1024 16-byte chunks per matrix; 256 threads x 4 chunks. Per wave this is
    // 8 async-load instructions -> ASYNCcnt += 8.
    auto stage = [&](int s, int buf) {
        const size_t kb = (size_t)s * KSTEP;
        #pragma unroll
        for (int c = 0; c < 4; ++c) {
            const int chunk = c * 256 + tid;       // 0..1023
            const int row   = chunk >> 3;          // 0..127
            const int col   = (chunk & 7) * 8;     // element offset in K-step
            const uint64_t ga = (uint64_t)(uintptr_t)
                (A + (size_t)(mBlk + row) * Kdim + kb + col);
            const uint32_t la = (uint32_t)(uintptr_t)
                (smem + (size_t)buf * 2 * TILE_ELEMS + row * LDAP + col);
            asm volatile("global_load_async_to_lds_b128 %0, %1, off"
                         :: "v"(la), "v"(ga) : "memory");
            const uint64_t gb = (uint64_t)(uintptr_t)
                (Bw + (size_t)(nBlk + row) * Kdim + kb + col);
            const uint32_t lb = (uint32_t)(uintptr_t)
                (smem + ((size_t)buf * 2 + 1) * TILE_ELEMS + row * LDAP + col);
            asm volatile("global_load_async_to_lds_b128 %0, %1, off"
                         :: "v"(lb), "v"(gb) : "memory");
        }
    };

    f32x8 acc[4][2];
    const f32x8 zero = {};
    #pragma unroll
    for (int i = 0; i < 4; ++i)
        #pragma unroll
        for (int j = 0; j < 2; ++j)
            acc[i][j] = zero;

    stage(0, 0);

    for (int s = 0; s < NS; ++s) {
        const int  buf  = s & 1;
        const bool more = (s + 1 < NS);
        if (more) stage(s + 1, buf ^ 1);           // prefetch next stage

        // Drain *previous* stage's 8 copies; the 8 just issued stay in flight.
        if (more) asm volatile("s_wait_asynccnt 0x8" ::: "memory");
        else      asm volatile("s_wait_asynccnt 0x0" ::: "memory");
        __syncthreads();                           // publish stage s to all waves

        const __bf16* sA = smem + (size_t)buf * 2 * TILE_ELEMS;
        const __bf16* sB = sA + TILE_ELEMS;

        #pragma unroll
        for (int ks = 0; ks < 2; ++ks) {           // two K=32 sub-steps
            const int k0 = ks * 32;
            FragAB a[4], b[2];
            #pragma unroll
            for (int i = 0; i < 4; ++i) {          // ds_load_b128 x2
                const __bf16* p = sA + (size_t)(mW + i * 16 + lr) * LDAP
                                     + k0 + lh * 8;
                a[i].u[0] = *(const u32x4*)(p);
                a[i].u[1] = *(const u32x4*)(p + 16);
            }
            #pragma unroll
            for (int j = 0; j < 2; ++j) {
                const __bf16* p = sB + (size_t)(nW + j * 16 + lr) * LDAP
                                     + k0 + lh * 8;
                b[j].u[0] = *(const u32x4*)(p);
                b[j].u[1] = *(const u32x4*)(p + 16);
            }
            #pragma unroll
            for (int i = 0; i < 4; ++i)
                #pragma unroll
                for (int j = 0; j < 2; ++j)
                    acc[i][j] = __builtin_amdgcn_wmma_f32_16x16x32_bf16(
                        false, a[i].v, false, b[j].v,
                        (short)0, acc[i][j], false, false);
        }
        __syncthreads();                           // done reading buf before refill
    }

    // Epilogue: C/D layout (7.12.2): VGPR r, lanes 0-15 -> M=base+r, N=lane;
    // lanes 16-31 -> M=base+8+r, N=lane-16.
    #pragma unroll
    for (int i = 0; i < 4; ++i) {
        #pragma unroll
        for (int j = 0; j < 2; ++j) {
            float* pc = C + (size_t)(mBlk + mW + i * 16 + lh * 8) * Ndim
                          + (nBlk + nW + j * 16 + lr);
            #pragma unroll
            for (int r = 0; r < 8; ++r) {
                float x = acc[i][j][r];
                if (act == 1) x = 1.0f / (1.0f + __expf(-x));
                pc[(size_t)r * Ndim] = x;
            }
        }
    }
}

// ---------------- kernel 4: WKV recurrence (serial in T per channel) --------
__global__ __launch_bounds__(256)
void wkv_scan(const float* __restrict__ kf,
              const float* __restrict__ vf,
              const float* __restrict__ rf,
              const float* __restrict__ time_decay,
              const float* __restrict__ time_first,
              __bf16* __restrict__ xo)
{
    const int a = blockIdx.x * blockDim.x + threadIdx.x;  // channel
    const int b = blockIdx.y;

    const float w  = -__expf(time_decay[a]);
    const float tf = time_first[a];

    float num = 0.f, den = 0.f, mx = -1e38f;
    size_t idx = ((size_t)b * Tsz) * Hsz + a;

    #pragma unroll 4
    for (int t = 0; t < Tsz; ++t, idx += Hsz) {
        const float kt = kf[idx];
        const float vt = vf[idx];
        const float mo  = fmaxf(mx, kt + tf);
        float e1 = __expf(mx - mo);
        float e2 = __expf(kt + tf - mo);
        const float out = (e1 * num + e2 * vt) / (e1 * den + e2);
        const float ms = fmaxf(mx + w, kt);
        e1 = __expf(mx + w - ms);
        e2 = __expf(kt - ms);
        num = e1 * num + e2 * vt;
        den = e1 * den + e2;
        mx  = ms;
        xo[idx] = (__bf16)(rf[idx] * out);
    }
}

// ---------------- launch ----------------------------------------------------
extern "C" void kernel_launch(void* const* d_in, const int* in_sizes, int n_in,
                              void* d_out, int out_size, void* d_ws, size_t ws_size,
                              hipStream_t stream)
{
    (void)in_sizes; (void)n_in; (void)out_size; (void)ws_size;

    const float* hidden = (const float*)d_in[0];
    const float* tdec   = (const float*)d_in[1];
    const float* tfir   = (const float*)d_in[2];
    const float* tmk    = (const float*)d_in[3];
    const float* tmv    = (const float*)d_in[4];
    const float* tmr    = (const float*)d_in[5];
    const float* Wk     = (const float*)d_in[6];
    const float* Wv     = (const float*)d_in[7];
    const float* Wr     = (const float*)d_in[8];
    const float* Wo     = (const float*)d_in[9];
    float* out = (float*)d_out;

    // workspace layout (total 192 MiB)
    char* ws = (char*)d_ws;
    const size_t actB = (size_t)Mrows * Hsz * sizeof(__bf16);  // 16 MiB
    const size_t wB   = (size_t)Hsz * Hsz * sizeof(__bf16);    //  8 MiB
    const size_t actF = (size_t)Mrows * Hsz * sizeof(float);   // 32 MiB

    __bf16* kin = (__bf16*)(ws);
    __bf16* vin = (__bf16*)(ws + actB);
    __bf16* rin = (__bf16*)(ws + 2 * actB);
    __bf16* Wkb = (__bf16*)(ws + 3 * actB);
    __bf16* Wvb = (__bf16*)(ws + 3 * actB + wB);
    __bf16* Wrb = (__bf16*)(ws + 3 * actB + 2 * wB);
    __bf16* Wob = (__bf16*)(ws + 3 * actB + 3 * wB);
    float*  kf  = (float*)(ws + 3 * actB + 4 * wB);
    float*  vf  = (float*)((char*)kf + actF);
    float*  rf  = (float*)((char*)vf + actF);
    __bf16* xo  = (__bf16*)((char*)rf + actF);

    // 1) time-mix prep
    prep_mix<<<dim3((Mrows * Hsz / 4) / 256), dim3(256), 0, stream>>>(
        hidden, tmk, tmv, tmr, kin, vin, rin);

    // 2) weight conversion
    const int cvtBlocks = (Hsz * Hsz / 4) / 256;
    cvt_f32_bf16<<<cvtBlocks, 256, 0, stream>>>(Wk, Wkb);
    cvt_f32_bf16<<<cvtBlocks, 256, 0, stream>>>(Wv, Wvb);
    cvt_f32_bf16<<<cvtBlocks, 256, 0, stream>>>(Wr, Wrb);
    cvt_f32_bf16<<<cvtBlocks, 256, 0, stream>>>(Wo, Wob);

    // 3) projection GEMMs (grid: N-tiles x M-tiles = 16 x 32)
    const dim3 gGemm(Hsz / 128, Mrows / 128);
    gemm_bf16<<<gGemm, 256, 0, stream>>>(kin, Wkb, kf, 0);
    gemm_bf16<<<gGemm, 256, 0, stream>>>(vin, Wvb, vf, 0);
    gemm_bf16<<<gGemm, 256, 0, stream>>>(rin, Wrb, rf, 1);  // fused sigmoid

    // 4) WKV scan (+ r-gating + bf16 cast)
    wkv_scan<<<dim3(Hsz / 256, Bsz), 256, 0, stream>>>(kf, vf, rf, tdec, tfir, xo);

    // 5) output projection GEMM -> fp32 d_out
    gemm_bf16<<<gGemm, 256, 0, stream>>>(xo, Wob, out, 0);
}
